// CoordGen_47519518163661
// MI455X (gfx1250) — matrix operand order repaired and checked by reading, not compiled, not run on hardware
//
#include <hip/hip_runtime.h>
#include <hip/hip_bf16.h>
#include <stdint.h>

// ---------------- problem constants (from reference) ----------------
constexpr int NB   = 128;      // graphs
constexpr int NA   = 100;      // atoms per graph
constexpr int NEPG = 2048;     // edges per graph
constexpr int NN   = NB * NA;       // 12800 nodes
constexpr int NE   = NB * NEPG;     // 262144 edges
constexpr int NLAT = 256;
constexpr int NHID = 128;
constexpr int NFCH = 512;
constexpr int NRBF = 16;
constexpr float EPSF   = 1e-8f;
constexpr float CUTOFF = 7.0f;

// MLP tiling: 64 edges (4 M-tiles of 16) per wave, 1 wave per workgroup.
constexpr int MT      = 4;                  // M-tiles per wave
constexpr int MROWS   = 16 * MT;            // 64 edge rows
constexpr int LDS_X2  = MROWS * 384;        // ushorts
constexpr int LDS_HH0 = MROWS * 512;        // ushorts
constexpr int LDS_BYTES = (LDS_X2 + LDS_HH0) * 2;   // 114688 B (<320KB WGP pool)

// ---------------- CDNA5 WMMA types ----------------
typedef __attribute__((ext_vector_type(16))) __bf16 v16bf;
typedef __attribute__((ext_vector_type(8)))  float  v8f;

union F16x16 { v16bf v; unsigned int u[8]; unsigned short h[16]; };

__device__ __forceinline__ unsigned short f2bf(float x) {
    unsigned int u = __float_as_uint(x);
    u += 0x7FFFu + ((u >> 16) & 1u);      // round-to-nearest-even
    return (unsigned short)(u >> 16);
}

// A-fragment (16x32 bf16, ISA 7.12.2 layout) loaded from an LDS row.
// row must already point at: base + row_m*stride + kbase + (lane>>4)*8
__device__ __forceinline__ v16bf load_a_lds(const unsigned short* row) {
    F16x16 r;
#pragma unroll
    for (int j = 0; j < 4; ++j) r.u[j] = *(const unsigned int*)(row + 2 * j);
#pragma unroll
    for (int j = 4; j < 8; ++j) r.u[j] = *(const unsigned int*)(row + 8 + 2 * j);
    return r.v;
}

__device__ __forceinline__ v8f wmma_bf16(v16bf a, v16bf b, v8f c) {
    return __builtin_amdgcn_wmma_f32_16x16x32_bf16(false, a, false, b, (short)0, c, false, false);
}

// ---------------- weight repack: f32 row-major [K][N] -> bf16 B-fragment order ----------------
// dst layout: [(kc*NT + nt)*32 + lane]*16 + i ; K_B(i,lane) = kc*32 + (i&7) + 16*(i>>3) + 8*(lane>>4), n = nt*16 + (lane&15)
__global__ void k_pack(const float* __restrict__ W, unsigned short* __restrict__ dst,
                       int K, int Nout, int Ksrc) {
    int idx = blockIdx.x * 256 + threadIdx.x;
    int total = (K >> 5) * (Nout >> 4) * 512;
    if (idx >= total) return;
    int i    = idx & 15;
    int lane = (idx >> 4) & 31;
    int rest = idx >> 9;
    int NT   = Nout >> 4;
    int nt   = rest % NT;
    int kc   = rest / NT;
    int n    = nt * 16 + (lane & 15);
    int Kidx = kc * 32 + (i & 7) + ((i >> 3) << 4) + ((lane >> 4) << 3);
    float v = (Kidx < Ksrc) ? W[Kidx * Nout + n] : 0.0f;
    dst[idx] = f2bf(v);
}

__global__ void k_lat16(const float* __restrict__ latents, unsigned short* __restrict__ lat16) {
    int idx = blockIdx.x * 256 + threadIdx.x;
    if (idx < NB * NLAT) lat16[idx] = f2bf(latents[idx]);
}

// ---------------- lattice + inverse + per-graph sigma ----------------
__global__ void k_lattice(const float* __restrict__ len, const float* __restrict__ ang,
                          const float* __restrict__ sigmas, const int* __restrict__ ts,
                          float* __restrict__ L, float* __restrict__ Li, float* __restrict__ sigB) {
    int b = threadIdx.x;
    if (b >= NB) return;
    const float D2R = 0.017453292519943295f;
    float a  = len[b * 3 + 0], bb = len[b * 3 + 1], c = len[b * 3 + 2];
    float al = ang[b * 3 + 0] * D2R, be = ang[b * 3 + 1] * D2R, ga = ang[b * 3 + 2] * D2R;
    float val = (cosf(al) * cosf(be) - cosf(ga)) / (sinf(al) * sinf(be));
    val = fminf(1.0f, fmaxf(-1.0f, val));
    float gs = acosf(val);
    float M[9];
    M[0] = a * sinf(be);              M[1] = 0.0f;                      M[2] = a * cosf(be);
    M[3] = -bb * sinf(al) * cosf(gs); M[4] = bb * sinf(al) * sinf(gs);  M[5] = bb * cosf(al);
    M[6] = 0.0f;                      M[7] = 0.0f;                      M[8] = c;
    float det = M[0] * (M[4] * M[8] - M[5] * M[7])
              - M[1] * (M[3] * M[8] - M[5] * M[6])
              + M[2] * (M[3] * M[7] - M[4] * M[6]);
    float id = 1.0f / det;
    float I[9];
    I[0] = (M[4] * M[8] - M[5] * M[7]) * id;
    I[1] = (M[2] * M[7] - M[1] * M[8]) * id;
    I[2] = (M[1] * M[5] - M[2] * M[4]) * id;
    I[3] = (M[5] * M[6] - M[3] * M[8]) * id;
    I[4] = (M[0] * M[8] - M[2] * M[6]) * id;
    I[5] = (M[2] * M[3] - M[0] * M[5]) * id;
    I[6] = (M[3] * M[7] - M[4] * M[6]) * id;
    I[7] = (M[1] * M[6] - M[0] * M[7]) * id;
    I[8] = (M[0] * M[4] - M[1] * M[3]) * id;
#pragma unroll
    for (int k = 0; k < 9; ++k) { L[b * 9 + k] = M[k]; Li[b * 9 + k] = I[k]; }
    sigB[b] = sigmas[ts[b]];
}

// ---------------- per-node geometry ----------------
__global__ void k_nodes(const float* __restrict__ frac, const float* __restrict__ noise,
                        const float* __restrict__ L, const float* __restrict__ Li,
                        const float* __restrict__ sigB,
                        float* __restrict__ pert, float* __restrict__ alg) {
    int n = blockIdx.x * blockDim.x + threadIdx.x;
    if (n >= NN) return;
    int b = n / NA;
    const float* Lb  = L  + b * 9;
    const float* Lib = Li + b * 9;
    float f0 = frac[n * 3 + 0], f1 = frac[n * 3 + 1], f2 = frac[n * 3 + 2];
    float cart[3], p0[3], fp[3], pw[3], fg[3], fd[3];
#pragma unroll
    for (int j = 0; j < 3; ++j)
        cart[j] = f0 * Lb[0 * 3 + j] + f1 * Lb[1 * 3 + j] + f2 * Lb[2 * 3 + j];
    float sig = sigB[b];
#pragma unroll
    for (int j = 0; j < 3; ++j) p0[j] = cart[j] + sig * noise[n * 3 + j];
#pragma unroll
    for (int j = 0; j < 3; ++j) {
        float v = p0[0] * Lib[0 * 3 + j] + p0[1] * Lib[1 * 3 + j] + p0[2] * Lib[2 * 3 + j];
        fp[j] = v - floorf(v);
    }
#pragma unroll
    for (int j = 0; j < 3; ++j)
        pw[j] = fp[0] * Lb[0 * 3 + j] + fp[1] * Lb[1 * 3 + j] + fp[2] * Lb[2 * 3 + j];
#pragma unroll
    for (int j = 0; j < 3; ++j)
        fg[j] = cart[0] * Lib[0 * 3 + j] + cart[1] * Lib[1 * 3 + j] + cart[2] * Lib[2 * 3 + j];
#pragma unroll
    for (int j = 0; j < 3; ++j) { float d = fg[j] - fp[j]; fd[j] = d - rintf(d); }
#pragma unroll
    for (int j = 0; j < 3; ++j) {
        pert[n * 3 + j] = pw[j];
        alg[n * 3 + j]  = pw[j] + fd[0] * Lb[0 * 3 + j] + fd[1] * Lb[1 * 3 + j] + fd[2] * Lb[2 * 3 + j];
    }
}

// ---------------- per-edge geometry ----------------
__global__ void k_edges(const int* __restrict__ ei, const int* __restrict__ tji,
                        const float* __restrict__ L, const float* __restrict__ pert,
                        const float* __restrict__ alg,
                        float* __restrict__ dvec, float* __restrict__ dist, float* __restrict__ gts) {
    int e = blockIdx.x * blockDim.x + threadIdx.x;
    if (e >= NE) return;
    int jn = ei[e], iN = ei[NE + e];
    int b = jn / NA;
    const float* Lb = L + b * 9;
    float t0 = (float)tji[e * 3 + 0], t1 = (float)tji[e * 3 + 1], t2 = (float)tji[e * 3 + 2];
    float dd = 0.0f, gg = 0.0f;
#pragma unroll
    for (int j = 0; j < 3; ++j) {
        float off = t0 * Lb[0 * 3 + j] + t1 * Lb[1 * 3 + j] + t2 * Lb[2 * 3 + j];
        float dv = pert[iN * 3 + j] - pert[jn * 3 + j] - off;
        float gv = alg[iN * 3 + j]  - alg[jn * 3 + j]  - off;
        dvec[e * 3 + j] = dv;
        dd += dv * dv;
        gg += gv * gv;
    }
    float d = sqrtf(dd);
    dist[e] = d;
    gts[e]  = sqrtf(gg) - d;
}

// ---------------- fused WMMA edge MLP ----------------
// 1 wave / block; each wave owns 64 edges (4 M-tiles) and 112KB dynamic LDS:
//   X2  : 64 x 384 bf16 (ef | latents)  -- its head doubles as the atom-type
//                                          staging buffer during stage A1
//   hh0 : 64 x 512 bf16 (xbuf 64x128 aliases its head)
// Each 32B B-fragment load from L2 feeds FOUR WMMAs -> 4x arithmetic
// intensity vs an M=16 tiling; L2 weight traffic ~3.75 GB/call.
__global__ __launch_bounds__(32) void k_mlp(
    const float* __restrict__ dist, const int* __restrict__ ei, const int* __restrict__ atype,
    const float* __restrict__ emb, const float* __restrict__ rbf_b, const float* __restrict__ bb_b,
    const float* __restrict__ b0, const float* __restrict__ b1,
    const float* __restrict__ w2, const float* __restrict__ b2,
    const unsigned short* __restrict__ rbfP, const unsigned short* __restrict__ bbP,
    const unsigned short* __restrict__ w0P, const unsigned short* __restrict__ w1P,
    const unsigned short* __restrict__ lat16, float* __restrict__ scores) {

    extern __shared__ unsigned short smem[];
    const int lane = threadIdx.x & 31;
    const int half = lane >> 4;
    const int ln   = lane & 15;

    unsigned short* X2   = smem;                 // 64 x 384
    unsigned short* hh0  = smem + LDS_X2;        // 64 x 512
    unsigned short* xbuf = hh0;                  // 64 x 128, alias: dead once hh0 is produced
    int* tybuf = (int*)X2;                       // [0:64)=type[jn], [64:128)=type[iN]; dead after A1

    const int e0 = blockIdx.x * MROWS;
    const int b  = e0 / NEPG;                    // all 64 edges of a tile share one graph

    __builtin_prefetch(w0P, 0, 1);
    __builtin_prefetch(w1P, 0, 1);

    // stage 0: cooperative atom-type gather into LDS (X2 head, free until A2)
    for (int t = lane; t < MROWS; t += 32) {
        tybuf[t]         = atype[ei[e0 + t]];
        tybuf[MROWS + t] = atype[ei[NE + e0 + t]];
    }

    // per-M-tile distances + rbf A fragments
    float d[MT];
#pragma unroll
    for (int mt = 0; mt < MT; ++mt) d[mt] = dist[e0 + mt * 16 + ln];
    F16x16 arb[MT];
#pragma unroll
    for (int i = 0; i < 16; ++i) {
        int K = (i & 7) + ((i >> 3) << 4) + (half << 3);
        if (K < NRBF) {
            float mu = (CUTOFF / 15.0f) * (float)K;
#pragma unroll
            for (int mt = 0; mt < MT; ++mt) {
                float t = d[mt] - mu;
                arb[mt].h[i] = f2bf(__expf(-10.0f * t * t));
            }
        } else {
#pragma unroll
            for (int mt = 0; mt < MT; ++mt) arb[mt].h[i] = 0;
        }
    }
    __syncthreads();

    // ---- stage A1: x = rbf @ rbf_W + rbf_b + h[jn] + h[iN] -> xbuf (64x128) ----
    for (int nt = 0; nt < 8; ++nt) {
        v16bf bf = *(const v16bf*)(rbfP + (size_t)(nt * 32 + lane) * 16);
        v8f a[MT];
#pragma unroll
        for (int mt = 0; mt < MT; ++mt) { a[mt] = v8f{}; a[mt] = wmma_bf16(arb[mt].v, bf, a[mt]); }
        int n = ln + nt * 16;
        float bias = rbf_b[n];
#pragma unroll
        for (int mt = 0; mt < MT; ++mt)
#pragma unroll
            for (int r = 0; r < 8; ++r) {
                int m = mt * 16 + half * 8 + r;
                float v = a[mt][r] + bias + emb[tybuf[m] * NHID + n] + emb[tybuf[MROWS + m] * NHID + n];
                xbuf[m * NHID + n] = f2bf(v);
            }
    }
    __syncthreads();

    // ---- stage A2: ef = silu(x @ bb_W + bb_b) -> X2[:, 0:128] ----
    for (int nt = 0; nt < 8; ++nt) {
        v8f a[MT];
#pragma unroll
        for (int mt = 0; mt < MT; ++mt) a[mt] = v8f{};
#pragma unroll
        for (int kc = 0; kc < 4; ++kc) {
            v16bf bf = *(const v16bf*)(bbP + (size_t)((kc * 8 + nt) * 32 + lane) * 16);
#pragma unroll
            for (int mt = 0; mt < MT; ++mt) {
                v16bf fa = load_a_lds(xbuf + (mt * 16 + ln) * NHID + kc * 32 + half * 8);
                a[mt] = wmma_bf16(fa, bf, a[mt]);
            }
        }
        int n = ln + nt * 16;
        float bias = bb_b[n];
#pragma unroll
        for (int mt = 0; mt < MT; ++mt)
#pragma unroll
            for (int r = 0; r < 8; ++r) {
                int m = mt * 16 + half * 8 + r;
                float v = a[mt][r] + bias;
                X2[m * 384 + n] = f2bf(v / (1.0f + __expf(-v)));
            }
    }
    // latents broadcast into X2[:, 128:384] (same graph for the whole tile), 16B chunks
    {
        const uint4* src = (const uint4*)(lat16 + b * NLAT);
        for (int t = lane; t < MROWS * 32; t += 32) {
            int m = t >> 5, q = t & 31;
            *(uint4*)(X2 + m * 384 + 128 + q * 8) = src[q];
        }
    }
    __syncthreads();

    // ---- stage B: hh0 = relu(X2 @ W0 + b0) -> hh0 (64x512) ----
    for (int nt = 0; nt < 32; ++nt) {
        v8f a[MT];
#pragma unroll
        for (int mt = 0; mt < MT; ++mt) a[mt] = v8f{};
#pragma unroll
        for (int kc = 0; kc < 12; ++kc) {
            v16bf bf = *(const v16bf*)(w0P + (size_t)((kc * 32 + nt) * 32 + lane) * 16);
#pragma unroll
            for (int mt = 0; mt < MT; ++mt) {
                v16bf fa = load_a_lds(X2 + (mt * 16 + ln) * 384 + kc * 32 + half * 8);
                a[mt] = wmma_bf16(fa, bf, a[mt]);
            }
        }
        int n = ln + nt * 16;
        float bias = b0[n];
#pragma unroll
        for (int mt = 0; mt < MT; ++mt)
#pragma unroll
            for (int r = 0; r < 8; ++r) {
                int m = mt * 16 + half * 8 + r;
                float v = a[mt][r] + bias;
                hh0[m * NFCH + n] = f2bf(v > 0.0f ? v : 0.0f);
            }
    }
    __syncthreads();

    // ---- stage C: hh1 = relu(hh0 @ W1 + b1); score = hh1 @ W2 + b2 ----
    float accS[MT][8];
#pragma unroll
    for (int mt = 0; mt < MT; ++mt)
#pragma unroll
        for (int r = 0; r < 8; ++r) accS[mt][r] = 0.0f;

    for (int nt = 0; nt < 32; ++nt) {
        v8f a[MT];
#pragma unroll
        for (int mt = 0; mt < MT; ++mt) a[mt] = v8f{};
#pragma unroll
        for (int kc = 0; kc < 16; ++kc) {
            v16bf bf = *(const v16bf*)(w1P + (size_t)((kc * 32 + nt) * 32 + lane) * 16);
#pragma unroll
            for (int mt = 0; mt < MT; ++mt) {
                v16bf fa = load_a_lds(hh0 + (mt * 16 + ln) * NFCH + kc * 32 + half * 8);
                a[mt] = wmma_bf16(fa, bf, a[mt]);
            }
        }
        int n = ln + nt * 16;
        float bias = b1[n];
        float w    = w2[n];
#pragma unroll
        for (int mt = 0; mt < MT; ++mt)
#pragma unroll
            for (int r = 0; r < 8; ++r) {
                float v = a[mt][r] + bias;
                accS[mt][r] += (v > 0.0f ? v : 0.0f) * w;
            }
    }
#pragma unroll
    for (int mt = 0; mt < MT; ++mt)
#pragma unroll
        for (int r = 0; r < 8; ++r) {
            float s = accS[mt][r];
            s += __shfl_xor(s, 1, 16);
            s += __shfl_xor(s, 2, 16);
            s += __shfl_xor(s, 4, 16);
            s += __shfl_xor(s, 8, 16);
            accS[mt][r] = s;
        }
    if (ln == 0) {
        float bb2 = b2[0];
#pragma unroll
        for (int mt = 0; mt < MT; ++mt)
#pragma unroll
            for (int r = 0; r < 8; ++r)
                scores[e0 + mt * 16 + half * 8 + r] = accS[mt][r] + bb2;
    }
}

// ---------------- per-graph segment means -> node accumulators ----------------
__global__ __launch_bounds__(256) void k_seg(const int* __restrict__ ei,
                                             const float* __restrict__ scores, const float* __restrict__ gts,
                                             const float* __restrict__ dvec, const float* __restrict__ dist,
                                             float* __restrict__ sN, float* __restrict__ gN) {
    int b = blockIdx.x;
    __shared__ float acc[NA * 6];
    for (int t = threadIdx.x; t < NA * 6; t += 256) acc[t] = 0.0f;
    __syncthreads();
    int base = b * NEPG;
    for (int el = threadIdx.x; el < NEPG; el += 256) {
        int e  = base + el;
        int i0 = ei[NE + e], j0 = ei[e];
        bool head = (el == 0) || (ei[NE + e - 1] != i0) || (ei[e - 1] != j0);
        if (head) {
            float s0 = 0, s1 = 0, s2 = 0, g0 = 0, g1 = 0, g2 = 0;
            int cnt = 0, ee = e;
            while (ee < base + NEPG && ei[NE + ee] == i0 && ei[ee] == j0) {
                float inv = 1.0f / (dist[ee] + EPSF);
                float sc = scores[ee] * inv, gg = gts[ee] * inv;
                float dx = dvec[ee * 3], dy = dvec[ee * 3 + 1], dz = dvec[ee * 3 + 2];
                s0 += sc * dx; s1 += sc * dy; s2 += sc * dz;
                g0 += gg * dx; g1 += gg * dy; g2 += gg * dz;
                ++cnt; ++ee;
            }
            float ic = 1.0f / (float)cnt;
            int a = i0 - b * NA;
            atomicAdd(&acc[a * 6 + 0], s0 * ic);
            atomicAdd(&acc[a * 6 + 1], s1 * ic);
            atomicAdd(&acc[a * 6 + 2], s2 * ic);
            atomicAdd(&acc[a * 6 + 3], g0 * ic);
            atomicAdd(&acc[a * 6 + 4], g1 * ic);
            atomicAdd(&acc[a * 6 + 5], g2 * ic);
        }
    }
    __syncthreads();
    for (int t = threadIdx.x; t < NA * 3; t += 256) {
        int a = t / 3, k = t % 3;
        sN[(b * NA + a) * 3 + k] = acc[a * 6 + k];
        gN[(b * NA + a) * 3 + k] = acc[a * 6 + 3 + k];
    }
}

// ---------------- per-graph loss + center ----------------
__global__ __launch_bounds__(128) void k_nodeloss(const float* __restrict__ sN, const float* __restrict__ gN,
                                                  const float* __restrict__ pert, const float* __restrict__ sigB,
                                                  float* __restrict__ preds, float* __restrict__ lossPart,
                                                  float* __restrict__ center) {
    int b = blockIdx.x, t = threadIdx.x;
    __shared__ float r0[128], r1[128], r2[128], r3[128];
    float sq = 0, p0 = 0, p1 = 0, p2 = 0;
    if (t < NA) {
        int n = b * NA + t;
        float isig = 1.0f / sigB[b];
        float pr[3];
#pragma unroll
        for (int k = 0; k < 3; ++k) {
            float s = sN[n * 3 + k], g = gN[n * 3 + k];
            float dd = s - g * isig;
            sq += dd * dd;
            pr[k] = pert[n * 3 + k] + s * isig;
            preds[n * 3 + k] = pr[k];
        }
        p0 = pr[0]; p1 = pr[1]; p2 = pr[2];
    }
    r0[t] = sq; r1[t] = p0; r2[t] = p1; r3[t] = p2;
    __syncthreads();
    for (int off = 64; off > 0; off >>= 1) {
        if (t < off) { r0[t] += r0[t + off]; r1[t] += r1[t + off]; r2[t] += r2[t + off]; r3[t] += r3[t + off]; }
        __syncthreads();
    }
    if (t == 0) {
        lossPart[b] = r0[0] / (3.0f * NA);
        center[b * 3 + 0] = r1[0] / NA;
        center[b * 3 + 1] = r2[0] / NA;
        center[b * 3 + 2] = r3[0] / NA;
    }
}

// ---------------- global min/max of centered preds (stage 1) ----------------
__global__ __launch_bounds__(256) void k_minmax(const float* __restrict__ preds, const float* __restrict__ center,
                                                float* __restrict__ mm) {
    __shared__ float s[256];
    int t = threadIdx.x;
    int n = blockIdx.x * 256 + t;
    bool ok = n < NN;
    int b = ok ? n / NA : 0;
    float dc[3] = {0, 0, 0};
    if (ok) {
#pragma unroll
        for (int k = 0; k < 3; ++k) dc[k] = preds[n * 3 + k] - center[b * 3 + k];
    }
#pragma unroll
    for (int c = 0; c < 3; ++c) {
        s[t] = ok ? dc[c] : -3.0e38f;
        __syncthreads();
        for (int off = 128; off > 0; off >>= 1) { if (t < off) s[t] = fmaxf(s[t], s[t + off]); __syncthreads(); }
        if (t == 0) mm[blockIdx.x * 6 + c * 2] = s[0];
        __syncthreads();
        s[t] = ok ? dc[c] : 3.0e38f;
        __syncthreads();
        for (int off = 128; off > 0; off >>= 1) { if (t < off) s[t] = fminf(s[t], s[t + off]); __syncthreads(); }
        if (t == 0) mm[blockIdx.x * 6 + c * 2 + 1] = s[0];
        __syncthreads();
    }
}

// ---------------- final scalars ----------------
__global__ void k_final(const float* __restrict__ lossPart, const float* __restrict__ mm,
                        int nblk, float* __restrict__ out) {
    float sl = 0.0f;
    for (int b = 0; b < NB; ++b) sl += lossPart[b];
    sl /= (float)NB;
    float mx[3] = {-3.0e38f, -3.0e38f, -3.0e38f};
    float mn[3] = { 3.0e38f,  3.0e38f,  3.0e38f};
    for (int p = 0; p < nblk; ++p) {
        for (int c = 0; c < 3; ++c) {
            mx[c] = fmaxf(mx[c], mm[p * 6 + c * 2]);
            mn[c] = fminf(mn[c], mm[p * 6 + c * 2 + 1]);
        }
    }
    out[0] = sl;
    out[1] = fabsf(mx[0] + mn[0]) + fabsf(mx[1] + mn[1]) + fabsf(mx[2] + mn[2]);
}

// ---------------- host launcher ----------------
extern "C" void kernel_launch(void* const* d_in, const int* in_sizes, int n_in,
                              void* d_out, int out_size, void* d_ws, size_t ws_size,
                              hipStream_t stream) {
    (void)in_sizes; (void)n_in; (void)out_size; (void)ws_size;
    const float* latents = (const float*)d_in[0];
    const float* frac    = (const float*)d_in[1];
    const float* lengths = (const float*)d_in[2];
    const float* angles  = (const float*)d_in[3];
    const float* noise   = (const float*)d_in[4];
    const float* sigmas  = (const float*)d_in[5];
    const float* emb     = (const float*)d_in[6];
    const float* rbf_W   = (const float*)d_in[7];
    const float* rbf_b   = (const float*)d_in[8];
    const float* bb_W    = (const float*)d_in[9];
    const float* bb_b    = (const float*)d_in[10];
    const float* sc_W0   = (const float*)d_in[11];
    const float* sc_b0   = (const float*)d_in[12];
    const float* sc_W1   = (const float*)d_in[13];
    const float* sc_b1   = (const float*)d_in[14];
    const float* sc_W2   = (const float*)d_in[15];
    const float* sc_b2   = (const float*)d_in[16];
    const int*   atype   = (const int*)d_in[18];
    const int*   ei      = (const int*)d_in[20];
    const int*   tji     = (const int*)d_in[21];
    const int*   ts      = (const int*)d_in[22];
    float* out = (float*)d_out;

    // workspace carve-up (f32 region then 32B-aligned bf16 region)
    float* f = (float*)d_ws;
    float* Lw   = f; f += NB * 9;
    float* Liw  = f; f += NB * 9;
    float* sigB = f; f += NB;
    float* pert = f; f += (size_t)NN * 3;
    float* alg  = f; f += (size_t)NN * 3;
    float* dvec = f; f += (size_t)NE * 3;
    float* dist = f; f += (size_t)NE;
    float* gts  = f; f += (size_t)NE;
    float* scrs = f; f += (size_t)NE;
    float* sNw  = f; f += (size_t)NN * 3;
    float* gNw  = f; f += (size_t)NN * 3;
    float* lossPart = f; f += NB;
    float* center   = f; f += NB * 3;
    float* preds    = f; f += (size_t)NN * 3;
    float* mm       = f; f += 64 * 6;
    unsigned short* us = (unsigned short*)(((uintptr_t)f + 31) & ~(uintptr_t)31);
    unsigned short* rbfP = us; us += 32 * NHID;
    unsigned short* bbP  = us; us += NHID * NHID;
    unsigned short* w0P  = us; us += (NHID + NLAT) * NFCH;
    unsigned short* w1P  = us; us += NFCH * NFCH;
    unsigned short* lat16 = us; us += NB * NLAT;

    // weight repack + latents conversion
    k_pack<<<(32 * NHID + 255) / 256, 256, 0, stream>>>(rbf_W, rbfP, 32, NHID, NRBF);
    k_pack<<<(NHID * NHID + 255) / 256, 256, 0, stream>>>(bb_W, bbP, NHID, NHID, NHID);
    k_pack<<<((NHID + NLAT) * NFCH + 255) / 256, 256, 0, stream>>>(sc_W0, w0P, NHID + NLAT, NFCH, NHID + NLAT);
    k_pack<<<(NFCH * NFCH + 255) / 256, 256, 0, stream>>>(sc_W1, w1P, NFCH, NFCH, NFCH);
    k_lat16<<<(NB * NLAT + 255) / 256, 256, 0, stream>>>(latents, lat16);

    // geometry
    k_lattice<<<1, 128, 0, stream>>>(lengths, angles, sigmas, ts, Lw, Liw, sigB);
    k_nodes<<<(NN + 255) / 256, 256, 0, stream>>>(frac, noise, Lw, Liw, sigB, pert, alg);
    k_edges<<<(NE + 255) / 256, 256, 0, stream>>>(ei, tji, Lw, pert, alg, dvec, dist, gts);

    // fused WMMA MLP: 4096 tiles of 64 edges, 1 wave/block, 112KB dynamic LDS
    k_mlp<<<NE / MROWS, 32, LDS_BYTES, stream>>>(dist, ei, atype, emb, rbf_b, bb_b,
                                                 sc_b0, sc_b1, sc_W2, sc_b2,
                                                 rbfP, bbP, w0P, w1P, lat16, scrs);

    // reductions
    k_seg<<<NB, 256, 0, stream>>>(ei, scrs, gts, dvec, dist, sNw, gNw);
    k_nodeloss<<<NB, 128, 0, stream>>>(sNw, gNw, pert, sigB, preds, lossPart, center);
    int nblk = (NN + 255) / 256;
    k_minmax<<<nblk, 256, 0, stream>>>(preds, center, mm);
    k_final<<<1, 1, 0, stream>>>(lossPart, mm, nblk, out);
}
